// MultiHeadAttention_77790447665701
// MI455X (gfx1250) — compile-verified, matrix operand order checked by reading, not run on hardware
//
#include <hip/hip_runtime.h>

// ---------------- CDNA5 WMMA types ----------------
typedef __attribute__((ext_vector_type(16))) __bf16 v16bf;
typedef __attribute__((ext_vector_type(8)))  float  v8f;

union FragU {
    v16bf bf;
    unsigned int u[8];
};

static __device__ __forceinline__ v8f wmma_bf16(v16bf a, v16bf b, v8f c) {
    // D = A(16x32 bf16) * B(32x16 bf16) + C(16x16 f32)
    return __builtin_amdgcn_wmma_f32_16x16x32_bf16(
        /*neg_a=*/false, a, /*neg_b=*/false, b,
        /*c_mod=*/(short)0, c, /*reuse_a=*/false, /*reuse_b=*/false);
}

static __device__ __forceinline__ unsigned short f2bf(float f) {
    unsigned int u = __float_as_uint(f);
    u += 0x7FFFu + ((u >> 16) & 1u);   // round to nearest even
    return (unsigned short)(u >> 16);
}
static __device__ __forceinline__ unsigned int f2bf2(float lo, float hi) {
    return (unsigned int)f2bf(lo) | ((unsigned int)f2bf(hi) << 16);
}

// A/B fragment where per-lane row index is (lane&15) and the contraction K
// runs contiguously in memory with unit stride (row-major rows of length
// `stride` ushorts). Covers K = 0..31 relative to `base`.
static __device__ __forceinline__ v16bf frag_rowK(const unsigned short* base, int stride) {
    int lane = threadIdx.x & 31;
    int m = lane & 15, half = lane >> 4;
    const unsigned short* r = base + m * stride + (half << 3);
    FragU f;
#pragma unroll
    for (int p = 0; p < 8; ++p) {
        int k = ((p & 3) << 1) + ((p >> 2) << 4);
        f.u[p] = *(const unsigned int*)(r + k);   // two consecutive bf16 (K, K+1)
    }
    return f.bf;
}

// B fragment where the contraction K strides by `stride` ushorts and the
// per-lane column (n = lane&15) is contiguous (K-major source, e.g. V tile).
static __device__ __forceinline__ v16bf frag_colK(const unsigned short* base, int stride) {
    int lane = threadIdx.x & 31;
    int n = lane & 15, half = lane >> 4;
    FragU f;
#pragma unroll
    for (int p = 0; p < 8; ++p) {
        int k = ((p & 3) << 1) + ((p >> 2) << 4) + (half << 3);
        unsigned int lo = base[(size_t)k * stride + n];
        unsigned int hi = base[(size_t)(k + 1) * stride + n];
        f.u[p] = lo | (hi << 16);
    }
    return f.bf;
}

// ---------------- problem constants ----------------
#define BATCH 4
#define SEQ   2048
#define DIM   1024
#define HEADS 16
#define DK    64
#define MROWS (BATCH * SEQ)   // 8192

// =====================================================================
// GEMM: y[m,n] = sum_k A[m,k] * W[n,k]   (A: MROWSxDIM, W: DIMxDIM)
// EPI==0: write bf16 into (B,H,S,DK) head-major buffer (QKV projection)
// EPI==1: write f32 row-major (final output projection)
// AT = float (fp32 source) or unsigned short (bf16 source)
// Block: 256 threads = 8 waves; tile BM=128, BN=128, BK=32.
// Wave grid 4x2; each wave owns 32x64 = 2x4 WMMA tiles (8 WMMA / k-step).
// =====================================================================
template <typename AT, int EPI>
__global__ __launch_bounds__(256) void gemm_wmma(const AT* __restrict__ A,
                                                 const float* __restrict__ W,
                                                 void* __restrict__ dst) {
    __shared__ unsigned short sA[128 * 32];
    __shared__ unsigned short sB[128 * 32];

    const int tid  = threadIdx.x;
    const int lane = tid & 31;
    const int wave = tid >> 5;
    const int wm = wave >> 1;          // 0..3 (rows, 32 each)
    const int wn = wave & 1;           // 0..1 (cols, 64 each)
    const int m0 = blockIdx.x * 128;
    const int n0 = blockIdx.y * 128;

    v8f acc[2][4] = {};

    for (int k0 = 0; k0 < DIM; k0 += 32) {
        // ---- stage A tile (128 x 32) as bf16 ----
        if constexpr (sizeof(AT) == 4) {
#pragma unroll
            for (int i = 0; i < 4; ++i) {
                int idx = tid + i * 256;                // 0..1023 float4 slots
                int row = idx >> 3, c = (idx & 7) * 4;
                const float* src = (const float*)A + (size_t)(m0 + row) * DIM + k0 + c;
                if (k0 + 32 < DIM) __builtin_prefetch(src + 32, 0, 0);  // next K tile
                const float4 v = *(const float4*)src;
                uint2 pk;
                pk.x = f2bf2(v.x, v.y);
                pk.y = f2bf2(v.z, v.w);
                *(uint2*)(sA + row * 32 + c) = pk;
            }
        } else {
#pragma unroll
            for (int i = 0; i < 2; ++i) {
                int idx = tid + i * 256;                // 0..511 uint4 slots
                int row = idx >> 2, c = (idx & 3) * 8;
                const unsigned short* src = (const unsigned short*)A + (size_t)(m0 + row) * DIM + k0 + c;
                if (k0 + 32 < DIM) __builtin_prefetch(src + 32, 0, 0);
                *(uint4*)(sA + row * 32 + c) = *(const uint4*)src;
            }
        }
        // ---- stage W tile (128 x 32) as bf16 (row n, cols k contiguous) ----
#pragma unroll
        for (int i = 0; i < 4; ++i) {
            int idx = tid + i * 256;                    // 0..1023 float4 slots
            int row = idx >> 3, c = (idx & 7) * 4;
            const float* src = W + (size_t)(n0 + row) * DIM + k0 + c;
            if (k0 + 32 < DIM) __builtin_prefetch(src + 32, 0, 0);
            const float4 v = *(const float4*)src;
            uint2 pk;
            pk.x = f2bf2(v.x, v.y);
            pk.y = f2bf2(v.z, v.w);
            *(uint2*)(sB + row * 32 + c) = pk;
        }
        __syncthreads();

        v16bf aF[2], bF[4];
#pragma unroll
        for (int mi = 0; mi < 2; ++mi)
            aF[mi] = frag_rowK(sA + (wm * 32 + mi * 16) * 32, 32);
#pragma unroll
        for (int ni = 0; ni < 4; ++ni)
            bF[ni] = frag_rowK(sB + (wn * 64 + ni * 16) * 32, 32);
#pragma unroll
        for (int mi = 0; mi < 2; ++mi)
#pragma unroll
            for (int ni = 0; ni < 4; ++ni)
                acc[mi][ni] = wmma_bf16(aF[mi], bF[ni], acc[mi][ni]);
        __syncthreads();
    }

    // ---- epilogue ----
    const int half = lane >> 4, nn = lane & 15;
#pragma unroll
    for (int mi = 0; mi < 2; ++mi)
#pragma unroll
        for (int ni = 0; ni < 4; ++ni)
#pragma unroll
            for (int r = 0; r < 8; ++r) {
                int mg = m0 + wm * 32 + mi * 16 + r + 8 * half;
                int ng = n0 + wn * 64 + ni * 16 + nn;
                float val = acc[mi][ni][r];
                if constexpr (EPI == 0) {
                    // scatter into (B,H,S,DK) bf16 head-major
                    int b = mg >> 11, s = mg & (SEQ - 1);
                    int h = ng >> 6, dk = ng & (DK - 1);
                    ((unsigned short*)dst)[(((size_t)(b * HEADS + h) * SEQ + s) << 6) + dk] = f2bf(val);
                } else {
                    ((float*)dst)[(size_t)mg * DIM + ng] = val;
                }
            }
}

// =====================================================================
// Flash attention: one wave32 per (b, h, 16-row query block).
// q/k/v: bf16 (B,H,S,DK). out: bf16 (B,S,D) concat-head row-major.
// =====================================================================
__global__ __launch_bounds__(32) void attn_wmma(const unsigned short* __restrict__ q,
                                                const unsigned short* __restrict__ k,
                                                const unsigned short* __restrict__ v,
                                                const int* __restrict__ pad,
                                                unsigned short* __restrict__ o) {
    __shared__ unsigned short sQ[16 * 64];
    __shared__ unsigned short sK[32 * 64];
    __shared__ unsigned short sV[32 * 64];
    __shared__ unsigned short sP[16 * 32];

    const int idx = blockIdx.x;
    const int qb = idx & 127;            // query block (S/16 = 128)
    const int h  = (idx >> 7) & (HEADS - 1);
    const int b  = idx >> 11;
    const int lane = threadIdx.x;
    const int half = lane >> 4, nn = lane & 15;

    const size_t headoff = (size_t)(b * HEADS + h) * SEQ * DK;

    // stage Q tile (16 x 64) : 128 uint4
    {
        const uint4* src = (const uint4*)(q + headoff + (size_t)qb * 16 * DK);
        uint4* d = (uint4*)sQ;
#pragma unroll
        for (int i = 0; i < 4; ++i) d[lane + i * 32] = src[lane + i * 32];
    }
    __syncthreads();
    const v16bf aQ0 = frag_rowK(sQ, 64);        // K (=dk) 0..31
    const v16bf aQ1 = frag_rowK(sQ + 32, 64);   // K (=dk) 32..63

    v8f accO[4] = {};
    float mrow[8], lrow[8];
#pragma unroll
    for (int r = 0; r < 8; ++r) { mrow[r] = -1e30f; lrow[r] = 0.0f; }

    const int nkb = ((qb * 16 + 15) >> 5) + 1;  // causal key-block bound
    for (int kb = 0; kb < nkb; ++kb) {
        // stage K and V tiles (32 x 64 each) : 256 uint4 each
        {
            const uint4* ks = (const uint4*)(k + headoff + (size_t)kb * 32 * DK);
            const uint4* vs = (const uint4*)(v + headoff + (size_t)kb * 32 * DK);
            uint4* dk_ = (uint4*)sK;
            uint4* dv_ = (uint4*)sV;
#pragma unroll
            for (int i = 0; i < 8; ++i) {
                dk_[lane + i * 32] = ks[lane + i * 32];
                dv_[lane + i * 32] = vs[lane + i * 32];
                if (kb + 1 < nkb) {                 // prefetch next key block
                    __builtin_prefetch(ks + 128 + lane + i * 32, 0, 0);
                    __builtin_prefetch(vs + 128 + lane + i * 32, 0, 0);
                }
            }
        }
        __syncthreads();

        // ---- logits S = Q * K^T (two 16x16 key tiles) ----
        v8f s0 = {}, s1 = {};
        {
            v16bf b00 = frag_rowK(sK + 0 * 64, 64);
            v16bf b01 = frag_rowK(sK + 0 * 64 + 32, 64);
            v16bf b10 = frag_rowK(sK + 16 * 64, 64);
            v16bf b11 = frag_rowK(sK + 16 * 64 + 32, 64);
            s0 = wmma_bf16(aQ0, b00, s0);
            s0 = wmma_bf16(aQ1, b01, s0);
            s1 = wmma_bf16(aQ0, b10, s1);
            s1 = wmma_bf16(aQ1, b11, s1);
        }

        const int key0 = kb * 32 + nn;
        const int key1 = key0 + 16;
        const int pm0 = pad[b * SEQ + key0];
        const int pm1 = pad[b * SEQ + key1];
        const float sc = 0.125f;   // 1/sqrt(64)

        // ---- masked scale + online softmax (rows = r + 8*half) ----
#pragma unroll
        for (int r = 0; r < 8; ++r) {
            const int qrow = qb * 16 + r + 8 * half;
            float x0 = s0[r] * sc, x1 = s1[r] * sc;
            if (key0 > qrow || pm0 == 0) x0 = -1e30f;
            if (key1 > qrow || pm1 == 0) x1 = -1e30f;

            float mx = fmaxf(x0, x1);
#pragma unroll
            for (int off = 1; off < 16; off <<= 1)
                mx = fmaxf(mx, __shfl_xor(mx, off, 32));
            const float mnew = fmaxf(mrow[r], mx);
            const float alpha = __expf(mrow[r] - mnew);
            const float e0 = __expf(x0 - mnew);
            const float e1 = __expf(x1 - mnew);
            float rs = e0 + e1;
#pragma unroll
            for (int off = 1; off < 16; off <<= 1)
                rs += __shfl_xor(rs, off, 32);
            lrow[r] = lrow[r] * alpha + rs;
            mrow[r] = mnew;
#pragma unroll
            for (int nt = 0; nt < 4; ++nt) accO[nt][r] *= alpha;

            // P back to LDS (bf16) in 16x32 row-major for the A operand
            sP[(r + 8 * half) * 32 + nn] = f2bf(e0);
            sP[(r + 8 * half) * 32 + 16 + nn] = f2bf(e1);
        }
        __syncthreads();

        // ---- O += P * V (4 dk tiles of 16) ----
        const v16bf aP = frag_rowK(sP, 32);
#pragma unroll
        for (int nt = 0; nt < 4; ++nt) {
            v16bf bV = frag_colK(sV + nt * 16, 64);
            accO[nt] = wmma_bf16(aP, bV, accO[nt]);
        }
        __syncthreads();
    }

    // ---- write O / l  → bf16 (B,S,D) concat-head layout ----
#pragma unroll
    for (int nt = 0; nt < 4; ++nt)
#pragma unroll
        for (int r = 0; r < 8; ++r) {
            const int srow = qb * 16 + r + 8 * half;
            const int d = nt * 16 + nn;
            const float val = accO[nt][r] / lrow[r];
            o[(size_t)(b * SEQ + srow) * DIM + h * DK + d] = f2bf(val);
        }
}

// =====================================================================
extern "C" void kernel_launch(void* const* d_in, const int* in_sizes, int n_in,
                              void* d_out, int out_size, void* d_ws, size_t ws_size,
                              hipStream_t stream) {
    const float* x_query = (const float*)d_in[0];
    const float* x_key   = (const float*)d_in[1];
    const float* x_value = (const float*)d_in[2];
    const int*   padmask = (const int*)d_in[3];
    const float* Wq = (const float*)d_in[4];
    const float* Wk = (const float*)d_in[5];
    const float* Wv = (const float*)d_in[6];
    const float* Wo = (const float*)d_in[7];
    float* out = (float*)d_out;

    // workspace: 4 bf16 buffers of MROWS*DIM elements each (16 MB each)
    unsigned short* ws = (unsigned short*)d_ws;
    const size_t BUF = (size_t)MROWS * DIM;   // 8388608 elems
    unsigned short* qb = ws;
    unsigned short* kb = ws + BUF;
    unsigned short* vb = ws + 2 * BUF;
    unsigned short* ab = ws + 3 * BUF;

    dim3 ggrid(MROWS / 128, DIM / 128);       // 64 x 8
    gemm_wmma<float, 0><<<ggrid, 256, 0, stream>>>(x_query, Wq, qb);
    gemm_wmma<float, 0><<<ggrid, 256, 0, stream>>>(x_key,   Wk, kb);
    gemm_wmma<float, 0><<<ggrid, 256, 0, stream>>>(x_value, Wv, vb);

    attn_wmma<<<BATCH * HEADS * (SEQ / 16), 32, 0, stream>>>(qb, kb, vb, padmask, ab);

    gemm_wmma<unsigned short, 1><<<ggrid, 256, 0, stream>>>(ab, Wo, out);
}
